// SoftmaxAttention_15891378995448
// MI455X (gfx1250) — matrix operand order repaired and checked by reading, not compiled, not run on hardware
//
#include <hip/hip_runtime.h>

typedef __attribute__((ext_vector_type(16))) _Float16 v16h;
typedef __attribute__((ext_vector_type(8)))  _Float16 v8h;
typedef __attribute__((ext_vector_type(8)))  float    v8f;

constexpr int L_SEQ = 2048;
constexpr int D_HEAD = 64;
constexpr int NHEADS = 8;
constexpr int NBATCH = 2;
constexpr int BM = 128;              // query rows per block (16 per wave, 8 waves)
constexpr int BN = 32;               // keys per streamed chunk
constexpr int NCHUNK = L_SEQ / BN;
constexpr int RK = 72;               // padded K-tile row stride (halves) -> 144B, conflict-free
constexpr int RV = 40;               // padded Vt / P row stride (halves) -> 80B

__device__ __forceinline__ v16h ld_b16x16(const _Float16* p) {
    // two aligned 16B LDS loads -> one 16xf16 WMMA operand
    const v8h lo = *(const v8h*)p;
    const v8h hi = *(const v8h*)(p + 8);
    return __builtin_shufflevector(lo, hi, 0,1,2,3,4,5,6,7,8,9,10,11,12,13,14,15);
}

__device__ __forceinline__ v8f wmma16(v16h a, v16h b, v8f c) {
    return __builtin_amdgcn_wmma_f32_16x16x32_f16(false, a, false, b, (short)0, c,
                                                  false, false);
}

__global__ __launch_bounds__(256)
void gmm_attn_kernel(const float* __restrict__ Q, const float* __restrict__ K,
                     const float* __restrict__ V, const float* __restrict__ pi,
                     const float* __restrict__ mask, float* __restrict__ out) {
    const int tid    = threadIdx.x;
    const int lane   = tid & 31;
    const int wave   = tid >> 5;
    const int laneLo = lane & 15;
    const int laneHi = lane >> 4;     // 0 or 1

    const int bh    = blockIdx.y;     // b*H + h
    const int b     = bh >> 3;        // H == 8
    const int qbase = blockIdx.x * BM + wave * 16;

    const float* Qb = Q + ((size_t)bh * L_SEQ + qbase) * D_HEAD;
    const float* Kb = K + (size_t)bh * L_SEQ * (2 * D_HEAD);
    const float* Vb = V + (size_t)bh * L_SEQ * D_HEAD;
    const float* Mb = mask + (size_t)b * L_SEQ;

    __shared__ __align__(16) _Float16 sK1[BN * RK];
    __shared__ __align__(16) _Float16 sK2[BN * RK];
    __shared__ __align__(16) _Float16 sVt[D_HEAD * RV];   // V transposed: [d][key]
    __shared__ __align__(16) _Float16 sP[8][16 * RV];     // per-wave prob staging
    __shared__ float sC1[BN];
    __shared__ float sC2[BN];

    const float pi0 = fminf(fmaxf(pi[0], 0.f), 1.f);
    const float pi1 = fminf(fmaxf(pi[1], 0.f), 1.f);

    // ---- Q tile -> WMMA A layout (2 operands: d 0..31, 32..63); row norms ----
    v16h aQ[2];
    float qn_part = 0.f;
    const float* qrow = Qb + laneLo * D_HEAD;
#pragma unroll
    for (int op = 0; op < 2; ++op) {
#pragma unroll
        for (int g = 0; g < 2; ++g) {
            const int d0 = op * 32 + laneHi * 16 + g * 8;
            const float4 f0 = *(const float4*)(qrow + d0);
            const float4 f1 = *(const float4*)(qrow + d0 + 4);
            qn_part += f0.x*f0.x + f0.y*f0.y + f0.z*f0.z + f0.w*f0.w
                     + f1.x*f1.x + f1.y*f1.y + f1.z*f1.z + f1.w*f1.w;
            const int e0 = g * 8;
            aQ[op][e0+0] = (_Float16)f0.x; aQ[op][e0+1] = (_Float16)f0.y;
            aQ[op][e0+2] = (_Float16)f0.z; aQ[op][e0+3] = (_Float16)f0.w;
            aQ[op][e0+4] = (_Float16)f1.x; aQ[op][e0+5] = (_Float16)f1.y;
            aQ[op][e0+6] = (_Float16)f1.z; aQ[op][e0+7] = (_Float16)f1.w;
        }
    }
    const float qn_full = qn_part + __shfl_xor(qn_part, 16, 32);
    float qnC[8];
#pragma unroll
    for (int r = 0; r < 8; ++r)
        qnC[r] = __shfl(qn_full, r + 8 * laneHi, 32) * (1.f / 16.f);

    v16h ones;
#pragma unroll
    for (int e = 0; e < 16; ++e) ones[e] = (_Float16)1.0f;

    const v8f vzero = {0.f,0.f,0.f,0.f,0.f,0.f,0.f,0.f};
    v8f acc[4];
#pragma unroll
    for (int dt = 0; dt < 4; ++dt) acc[dt] = vzero;
    v8f accD = vzero;

    const int key  = tid >> 3;        // 0..31
    const int dseg = (tid & 7) * 8;   // 0..56

    for (int kc = 0; kc < NCHUNK; ++kc) {
        const int kbase = kc * BN;
        __syncthreads();
        { // ---- cooperative chunk load: f32 -> f16 into LDS ----
            const float* k1p = Kb + (size_t)(kbase + key) * (2 * D_HEAD) + dseg;
            const float* k2p = k1p + D_HEAD;
            const float* vp  = Vb + (size_t)(kbase + key) * D_HEAD + dseg;
            const float4 a0 = *(const float4*)(k1p);
            const float4 a1 = *(const float4*)(k1p + 4);
            const float4 b0 = *(const float4*)(k2p);
            const float4 b1 = *(const float4*)(k2p + 4);
            const float4 v0 = *(const float4*)(vp);
            const float4 v1 = *(const float4*)(vp + 4);
            if (kc + 1 < NCHUNK) { // global_prefetch_b8 for next chunk
                __builtin_prefetch(k1p + BN * 2 * D_HEAD, 0, 1);
                __builtin_prefetch(k2p + BN * 2 * D_HEAD, 0, 1);
                __builtin_prefetch(vp  + BN * D_HEAD, 0, 1);
            }
            v8h h1, h2;
            h1[0]=(_Float16)a0.x; h1[1]=(_Float16)a0.y; h1[2]=(_Float16)a0.z; h1[3]=(_Float16)a0.w;
            h1[4]=(_Float16)a1.x; h1[5]=(_Float16)a1.y; h1[6]=(_Float16)a1.z; h1[7]=(_Float16)a1.w;
            h2[0]=(_Float16)b0.x; h2[1]=(_Float16)b0.y; h2[2]=(_Float16)b0.z; h2[3]=(_Float16)b0.w;
            h2[4]=(_Float16)b1.x; h2[5]=(_Float16)b1.y; h2[6]=(_Float16)b1.z; h2[7]=(_Float16)b1.w;
            *(v8h*)&sK1[key * RK + dseg] = h1;
            *(v8h*)&sK2[key * RK + dseg] = h2;
            sVt[(dseg+0)*RV + key] = (_Float16)v0.x;
            sVt[(dseg+1)*RV + key] = (_Float16)v0.y;
            sVt[(dseg+2)*RV + key] = (_Float16)v0.z;
            sVt[(dseg+3)*RV + key] = (_Float16)v0.w;
            sVt[(dseg+4)*RV + key] = (_Float16)v1.x;
            sVt[(dseg+5)*RV + key] = (_Float16)v1.y;
            sVt[(dseg+6)*RV + key] = (_Float16)v1.z;
            sVt[(dseg+7)*RV + key] = (_Float16)v1.w;
            float pk1 = a0.x*a0.x + a0.y*a0.y + a0.z*a0.z + a0.w*a0.w
                      + a1.x*a1.x + a1.y*a1.y + a1.z*a1.z + a1.w*a1.w;
            float pk2 = b0.x*b0.x + b0.y*b0.y + b0.z*b0.z + b0.w*b0.w
                      + b1.x*b1.x + b1.y*b1.y + b1.z*b1.z + b1.w*b1.w;
            pk1 += __shfl_xor(pk1, 1, 32); pk1 += __shfl_xor(pk1, 2, 32); pk1 += __shfl_xor(pk1, 4, 32);
            pk2 += __shfl_xor(pk2, 1, 32); pk2 += __shfl_xor(pk2, 2, 32); pk2 += __shfl_xor(pk2, 4, 32);
            if ((tid & 7) == 0) {
                const float mb = 1.0e6f * (1.f - Mb[kbase + key]);
                sC1[key] = pk1 * (1.f / 16.f) + mb;
                sC2[key] = pk2 * (1.f / 16.f) + mb;
            }
        }
        __syncthreads();

        // ---- per-wave: scores (2x 16x16 tiles), mixture weights -> sP ----
        _Float16* pw = &sP[wave][0];
#pragma unroll
        for (int nt = 0; nt < 2; ++nt) {
            const _Float16* r1 = &sK1[(nt * 16 + laneLo) * RK];
            const _Float16* r2 = &sK2[(nt * 16 + laneLo) * RK];
            v8f c1 = vzero, c2 = vzero;
            c1 = wmma16(aQ[0], ld_b16x16(r1 + laneHi * 16), c1);
            c1 = wmma16(aQ[1], ld_b16x16(r1 + 32 + laneHi * 16), c1);
            c2 = wmma16(aQ[0], ld_b16x16(r2 + laneHi * 16), c2);
            c2 = wmma16(aQ[1], ld_b16x16(r2 + 32 + laneHi * 16), c2);
            const float cc1 = sC1[nt * 16 + laneLo];
            const float cc2 = sC2[nt * 16 + laneLo];
#pragma unroll
            for (int r = 0; r < 8; ++r) {
                const float s1 = c1[r] * 0.125f - qnC[r] - cc1;
                const float s2 = c2[r] * 0.125f - qnC[r] - cc2;
                const float w  = pi0 * __expf(s1) + pi1 * __expf(s2);
                pw[(r + 8 * laneHi) * RV + nt * 16 + laneLo] = (_Float16)w;
            }
        }
        asm volatile("s_wait_dscnt 0" ::: "memory"); // same-wave LDS RAW on sP

        // ---- read probs back in A layout; denominator + PV WMMAs ----
        const _Float16* pr = pw + laneLo * RV;
        const v8h plo = *(const v8h*)(pr + laneHi * 16);
        const v8h phi = *(const v8h*)(pr + laneHi * 16 + 8);
        const v16h aP = __builtin_shufflevector(plo, phi,
                          0,1,2,3,4,5,6,7,8,9,10,11,12,13,14,15);
        accD = wmma16(aP, ones, accD);   // rowsum(w), replicated over lanes
#pragma unroll
        for (int dt = 0; dt < 4; ++dt) {
            const v16h bV = ld_b16x16(&sVt[(dt * 16 + laneLo) * RV + laneHi * 16]);
            acc[dt] = wmma16(aP, bV, acc[dt]);
        }
    }

    // ---- normalize and store (C layout: row = r+8*laneHi, col = dt*16+laneLo) ----
    float* ob = out + ((size_t)bh * L_SEQ + qbase) * D_HEAD;
#pragma unroll
    for (int r = 0; r < 8; ++r) {
        const float inv = 1.f / accD[r];
        const int row = r + 8 * laneHi;
#pragma unroll
        for (int dt = 0; dt < 4; ++dt)
            ob[row * D_HEAD + dt * 16 + laneLo] = acc[dt][r] * inv;
    }
}

extern "C" void kernel_launch(void* const* d_in, const int* in_sizes, int n_in,
                              void* d_out, int out_size, void* d_ws, size_t ws_size,
                              hipStream_t stream) {
    (void)in_sizes; (void)n_in; (void)out_size; (void)d_ws; (void)ws_size;
    const float* Q    = (const float*)d_in[0];
    const float* K    = (const float*)d_in[1];
    const float* V    = (const float*)d_in[2];
    const float* pi   = (const float*)d_in[3];
    const float* mask = (const float*)d_in[4];
    float* out = (float*)d_out;
    dim3 grid(L_SEQ / BM, NBATCH * NHEADS);
    gmm_attn_kernel<<<grid, 256, 0, stream>>>(Q, K, V, pi, mask, out);
}
